// ODE_RNN_Model_76493367542058
// MI455X (gfx1250) — matrix operand order repaired and checked by reading, not compiled, not run on hardware
//
#include <hip/hip_runtime.h>

typedef __attribute__((ext_vector_type(16))) _Float16 v16h;
typedef __attribute__((ext_vector_type(8)))  _Float16 v8h;
typedef __attribute__((ext_vector_type(8)))  float    v8f;

namespace {

constexpr int kB     = 20000;
constexpr int kTobs  = 40;
constexpr int kTtgt  = 10;
constexpr int kL     = 64;
constexpr int kNH    = 128;
constexpr int kG     = 192;
constexpr int kRK4   = 20;
constexpr int kWaves = 8;     // 8 waves/WG -> 2 waves per SIMD32, 157 WGs total
constexpr int ST64   = 72;    // padded column stride (f16 elems) for K=64 weights
constexpr int ST128  = 136;   // padded column stride (f16 elems) for K=128 weights
constexpr int SROW   = 136;   // per-wave staging row stride (f16 elems)

// ---------------------------------------------------------------- WMMA helpers

__device__ inline v8f wmma_f16(v16h a, v16h b, v8f c) {
  return __builtin_amdgcn_wmma_f32_16x16x32_f16(false, a, false, b, (short)0, c,
                                                false, false);
}

__device__ inline float relu(float x) { return (x > 0.f) ? x : 0.f; }

// A fragment (16x32 f16) from row-major staging. Per ISA layout:
// lane<16 holds row=lane, K = {half*8..half*8+7, 16+half*8..16+half*8+7}
__device__ inline v16h load_a_frag(const _Float16* s, int lid, int kbase) {
  const int half = lid >> 4;
  const int row  = lid & 15;
  const _Float16* p = s + row * SROW + kbase + half * 8;
  v8h c0 = *(const v8h*)(p);
  v8h c1 = *(const v8h*)(p + 16);
  v16h a;
#pragma unroll
  for (int i = 0; i < 8; ++i) { a[i] = c0[i]; a[8 + i] = c1[i]; }
  return a;
}

// B fragment (32x16 f16) from column-major (N-major) weight tile in LDS.
// lane holds column N = lane&15, K = {half*8..half*8+7, 16+half*8..16+half*8+7}.
__device__ inline v16h load_b_frag(const _Float16* w, int stride, int lid,
                                   int nbase, int kbase) {
  const int half = lid >> 4;
  const int col  = lid & 15;
  const _Float16* p = w + (nbase + col) * stride + kbase + half * 8;
  v8h c0 = *(const v8h*)(p);
  v8h c1 = *(const v8h*)(p + 16);
  v16h b;
#pragma unroll
  for (int i = 0; i < 8; ++i) { b[i] = c0[i]; b[8 + i] = c1[i]; }
  return b;
}

__device__ inline v8f bias_tile(const float* b, int lid, int nbase) {
  const float v = b[nbase + (lid & 15)];
  v8f c;
#pragma unroll
  for (int r = 0; r < 8; ++r) c[r] = v;
  return c;
}

__device__ inline v8f bias_tile2(const float* b0, const float* b1, int lid, int nbase) {
  const int n = nbase + (lid & 15);
  const float v = b0[n] + b1[n];
  v8f c;
#pragma unroll
  for (int r = 0; r < 8; ++r) c[r] = v;
  return c;
}

// Store a 16x16 f32 C tile into f16 row-major staging.
// C layout: VGPR r -> row = (lane>=16 ? 8 : 0) + r, col = lane&15.
__device__ inline void store_c_f16(_Float16* s, int lid, int colbase, v8f c) {
  const int col   = colbase + (lid & 15);
  const int rbase = (lid >> 4) * 8;
#pragma unroll
  for (int r = 0; r < 8; ++r) s[(rbase + r) * SROW + col] = (_Float16)c[r];
}

__device__ inline float fsigmoid(float x) { return 1.f / (1.f + __expf(-x)); }
__device__ inline float ftanh(float x) { return 1.f - 2.f / (__expf(2.f * x) + 1.f); }

// ---------------------------------------------------------------- ODE MLP

struct OdeW {
  const _Float16 *W1, *Wm, *Wf;
  const float *b1, *bm, *bf;
};

// Input (16x64 f16) already staged at stg cols 0..63. out = f(input), C-layout.
__device__ inline void ode_f(_Float16* stg, const OdeW& w, int lid, v8f out[4]) {
  v16h a0 = load_a_frag(stg, lid, 0);
  v16h a1 = load_a_frag(stg, lid, 32);
#pragma unroll
  for (int nt = 0; nt < 8; ++nt) {               // layer1: 16x64 @ 64x128
    v8f c = bias_tile(w.b1, lid, nt * 16);
    c = wmma_f16(a0, load_b_frag(w.W1, ST64, lid, nt * 16, 0),  c);
    c = wmma_f16(a1, load_b_frag(w.W1, ST64, lid, nt * 16, 32), c);
#pragma unroll
    for (int r = 0; r < 8; ++r) c[r] = relu(c[r]);
    store_c_f16(stg, lid, nt * 16, c);
  }
  v16h m0 = load_a_frag(stg, lid, 0);
  v16h m1 = load_a_frag(stg, lid, 32);
  v16h m2 = load_a_frag(stg, lid, 64);
  v16h m3 = load_a_frag(stg, lid, 96);
#pragma unroll
  for (int nt = 0; nt < 8; ++nt) {               // layer2: 16x128 @ 128x128
    v8f c = bias_tile(w.bm, lid, nt * 16);
    c = wmma_f16(m0, load_b_frag(w.Wm, ST128, lid, nt * 16, 0),  c);
    c = wmma_f16(m1, load_b_frag(w.Wm, ST128, lid, nt * 16, 32), c);
    c = wmma_f16(m2, load_b_frag(w.Wm, ST128, lid, nt * 16, 64), c);
    c = wmma_f16(m3, load_b_frag(w.Wm, ST128, lid, nt * 16, 96), c);
#pragma unroll
    for (int r = 0; r < 8; ++r) c[r] = relu(c[r]);
    store_c_f16(stg, lid, nt * 16, c);
  }
  v16h f0 = load_a_frag(stg, lid, 0);
  v16h f1 = load_a_frag(stg, lid, 32);
  v16h f2 = load_a_frag(stg, lid, 64);
  v16h f3 = load_a_frag(stg, lid, 96);
#pragma unroll
  for (int nt = 0; nt < 4; ++nt) {               // layer3: 16x128 @ 128x64
    v8f c = bias_tile(w.bf, lid, nt * 16);
    c = wmma_f16(f0, load_b_frag(w.Wf, ST128, lid, nt * 16, 0),  c);
    c = wmma_f16(f1, load_b_frag(w.Wf, ST128, lid, nt * 16, 32), c);
    c = wmma_f16(f2, load_b_frag(w.Wf, ST128, lid, nt * 16, 64), c);
    c = wmma_f16(f3, load_b_frag(w.Wf, ST128, lid, nt * 16, 96), c);
    out[nt] = c;
  }
}

__device__ inline void stage_h(_Float16* stg, int lid, const v8f h[4], float s,
                               const v8f k[4]) {
#pragma unroll
  for (int nt = 0; nt < 4; ++nt) {
    v8f x;
#pragma unroll
    for (int r = 0; r < 8; ++r) x[r] = h[nt][r] + s * k[nt][r];
    store_c_f16(stg, lid, nt * 16, x);
  }
}

__device__ inline void rk4_evolve(v8f h[4], float dt, _Float16* stg,
                                  const OdeW& w, int lid) {
  const float step = dt * (1.f / kRK4);
#pragma unroll 1
  for (int it = 0; it < kRK4; ++it) {
    v8f k[4], acc[4];
#pragma unroll
    for (int nt = 0; nt < 4; ++nt) store_c_f16(stg, lid, nt * 16, h[nt]);
    ode_f(stg, w, lid, k);                                   // k1
#pragma unroll
    for (int nt = 0; nt < 4; ++nt)
#pragma unroll
      for (int r = 0; r < 8; ++r) acc[nt][r] = h[nt][r] + (step / 6.f) * k[nt][r];
    stage_h(stg, lid, h, 0.5f * step, k);
    ode_f(stg, w, lid, k);                                   // k2
#pragma unroll
    for (int nt = 0; nt < 4; ++nt)
#pragma unroll
      for (int r = 0; r < 8; ++r) acc[nt][r] += (step / 3.f) * k[nt][r];
    stage_h(stg, lid, h, 0.5f * step, k);
    ode_f(stg, w, lid, k);                                   // k3
#pragma unroll
    for (int nt = 0; nt < 4; ++nt)
#pragma unroll
      for (int r = 0; r < 8; ++r) acc[nt][r] += (step / 3.f) * k[nt][r];
    stage_h(stg, lid, h, step, k);
    ode_f(stg, w, lid, k);                                   // k4
#pragma unroll
    for (int nt = 0; nt < 4; ++nt)
#pragma unroll
      for (int r = 0; r < 8; ++r) h[nt][r] = acc[nt][r] + (step / 6.f) * k[nt][r];
  }
}

// ---------------------------------------------------------------- GRU cell
// stg cols 0..63 hold x_emb (f16) on entry; h staged to cols 64..127 here.
__device__ inline void gru_step(v8f h[4], _Float16* stg,
                                const _Float16* sWih, const _Float16* sWhh,
                                const float* sbih, const float* sbhh, int lid) {
#pragma unroll
  for (int nt = 0; nt < 4; ++nt) store_c_f16(stg, lid, 64 + nt * 16, h[nt]);
  v16h x0 = load_a_frag(stg, lid, 0);
  v16h x1 = load_a_frag(stg, lid, 32);
  v16h g0 = load_a_frag(stg, lid, 64);
  v16h g1 = load_a_frag(stg, lid, 96);
#pragma unroll
  for (int nt = 0; nt < 4; ++nt) {
    const int nr = nt * 16, nz = 64 + nt * 16, nn = 128 + nt * 16;
    v8f cr = bias_tile2(sbih, sbhh, lid, nr);
    cr = wmma_f16(x0, load_b_frag(sWih, ST64, lid, nr, 0),  cr);
    cr = wmma_f16(x1, load_b_frag(sWih, ST64, lid, nr, 32), cr);
    cr = wmma_f16(g0, load_b_frag(sWhh, ST64, lid, nr, 0),  cr);
    cr = wmma_f16(g1, load_b_frag(sWhh, ST64, lid, nr, 32), cr);
    v8f cz = bias_tile2(sbih, sbhh, lid, nz);
    cz = wmma_f16(x0, load_b_frag(sWih, ST64, lid, nz, 0),  cz);
    cz = wmma_f16(x1, load_b_frag(sWih, ST64, lid, nz, 32), cz);
    cz = wmma_f16(g0, load_b_frag(sWhh, ST64, lid, nz, 0),  cz);
    cz = wmma_f16(g1, load_b_frag(sWhh, ST64, lid, nz, 32), cz);
    v8f ci = bias_tile(sbih, lid, nn);
    ci = wmma_f16(x0, load_b_frag(sWih, ST64, lid, nn, 0),  ci);
    ci = wmma_f16(x1, load_b_frag(sWih, ST64, lid, nn, 32), ci);
    v8f ch = bias_tile(sbhh, lid, nn);
    ch = wmma_f16(g0, load_b_frag(sWhh, ST64, lid, nn, 0),  ch);
    ch = wmma_f16(g1, load_b_frag(sWhh, ST64, lid, nn, 32), ch);
#pragma unroll
    for (int r = 0; r < 8; ++r) {
      const float rg = fsigmoid(cr[r]);
      const float zg = fsigmoid(cz[r]);
      const float ng = ftanh(ci[r] + rg * ch[r]);
      h[nt][r] = (1.f - zg) * ng + zg * h[nt][r];
    }
  }
}

// ---------------------------------------------------------------- decoder
__device__ inline void decode_emit(const v8f h[4], _Float16* stg,
                                   const _Float16* sD1, const _Float16* sD2,
                                   const _Float16* sD3, const float* sdb1,
                                   const float* sdb2, const float* sdb3, int lid,
                                   int b0, int t, float* __restrict__ pred,
                                   const float* __restrict__ obs,
                                   const float* __restrict__ tgt,
                                   float& accR, float& accP) {
#pragma unroll
  for (int nt = 0; nt < 4; ++nt) store_c_f16(stg, lid, nt * 16, h[nt]);
  v16h a0 = load_a_frag(stg, lid, 0);
  v16h a1 = load_a_frag(stg, lid, 32);
#pragma unroll
  for (int nt = 0; nt < 8; ++nt) {
    v8f c = bias_tile(sdb1, lid, nt * 16);
    c = wmma_f16(a0, load_b_frag(sD1, ST64, lid, nt * 16, 0),  c);
    c = wmma_f16(a1, load_b_frag(sD1, ST64, lid, nt * 16, 32), c);
#pragma unroll
    for (int r = 0; r < 8; ++r) c[r] = relu(c[r]);
    store_c_f16(stg, lid, nt * 16, c);
  }
  v16h m0 = load_a_frag(stg, lid, 0);
  v16h m1 = load_a_frag(stg, lid, 32);
  v16h m2 = load_a_frag(stg, lid, 64);
  v16h m3 = load_a_frag(stg, lid, 96);
#pragma unroll
  for (int nt = 0; nt < 8; ++nt) {
    v8f c = bias_tile(sdb2, lid, nt * 16);
    c = wmma_f16(m0, load_b_frag(sD2, ST128, lid, nt * 16, 0),  c);
    c = wmma_f16(m1, load_b_frag(sD2, ST128, lid, nt * 16, 32), c);
    c = wmma_f16(m2, load_b_frag(sD2, ST128, lid, nt * 16, 64), c);
    c = wmma_f16(m3, load_b_frag(sD2, ST128, lid, nt * 16, 96), c);
#pragma unroll
    for (int r = 0; r < 8; ++r) c[r] = relu(c[r]);
    store_c_f16(stg, lid, nt * 16, c);
  }
  v16h f0 = load_a_frag(stg, lid, 0);
  v16h f1 = load_a_frag(stg, lid, 32);
  v16h f2 = load_a_frag(stg, lid, 64);
  v16h f3 = load_a_frag(stg, lid, 96);
  v8f c = bias_tile(sdb3, lid, 0);            // cols 3..15 are zero-padded
  c = wmma_f16(f0, load_b_frag(sD3, ST128, lid, 0, 0),  c);
  c = wmma_f16(f1, load_b_frag(sD3, ST128, lid, 0, 32), c);
  c = wmma_f16(f2, load_b_frag(sD3, ST128, lid, 0, 64), c);
  c = wmma_f16(f3, load_b_frag(sD3, ST128, lid, 0, 96), c);
  const int col   = lid & 15;
  const int rbase = (lid >> 4) * 8;
  if (col < 3) {
#pragma unroll
    for (int r = 0; r < 8; ++r) {
      const int row = b0 + rbase + r;
      const float p = c[r];
      pred[row * (kTobs + kTtgt) * 3 + t * 3 + col] = p;
      float ref, e;
      if (t < kTobs) {
        ref = obs[row * kTobs * 3 + t * 3 + col];
        e = p - ref; accR += e * e;
      } else {
        ref = tgt[row * kTtgt * 3 + (t - kTobs) * 3 + col];
        e = p - ref; accP += e * e;
      }
    }
  }
}

// ---------------------------------------------------------------- weight stage
__device__ inline void stage_w(_Float16* dst, const float* src, int nout, int kin,
                               int stride, int tid, int nth) {
  for (int i = tid; i < nout * kin; i += nth) {
    const int n = i / kin, k = i - n * kin;
    dst[n * stride + k] = (_Float16)src[i];   // src is (out,in) row-major = B^T col
  }
}

// ---------------------------------------------------------------- main kernel
__global__ __launch_bounds__(kWaves * 32, 2)
void ode_rnn_fused(const float* __restrict__ obs, const float* __restrict__ tgt,
                   const float* __restrict__ ft, const float* __restrict__ embW,
                   const float* __restrict__ embB, const float* __restrict__ Wih,
                   const float* __restrict__ Whh, const float* __restrict__ bih,
                   const float* __restrict__ bhh, const float* __restrict__ W1,
                   const float* __restrict__ b1, const float* __restrict__ Wm,
                   const float* __restrict__ bm, const float* __restrict__ Wf,
                   const float* __restrict__ bf, const float* __restrict__ dW1,
                   const float* __restrict__ db1, const float* __restrict__ dW2,
                   const float* __restrict__ db2, const float* __restrict__ dW3,
                   const float* __restrict__ db3, float* __restrict__ out,
                   float* __restrict__ ws) {
  __shared__ __align__(16) _Float16 sWih[kG * ST64];
  __shared__ __align__(16) _Float16 sWhh[kG * ST64];
  __shared__ __align__(16) _Float16 sW1[kNH * ST64];
  __shared__ __align__(16) _Float16 sWm[kNH * ST128];
  __shared__ __align__(16) _Float16 sWf[kL * ST128];
  __shared__ __align__(16) _Float16 sD1[kNH * ST64];
  __shared__ __align__(16) _Float16 sD2[kNH * ST128];
  __shared__ __align__(16) _Float16 sD3[16 * ST128];
  __shared__ float sb1[kNH], sbm[kNH], sbf[kL], sbih[kG], sbhh[kG];
  __shared__ float sdb1[kNH], sdb2[kNH], sdb3[16];
  __shared__ float sembW[kL * 3], sembB[kL];
  __shared__ __align__(16) _Float16 stgAll[kWaves * 16 * SROW];
  __shared__ float obsrAll[kWaves * 16 * 3];

  const int tid = threadIdx.x, nth = blockDim.x;
  stage_w(sWih, Wih, kG, kL, ST64, tid, nth);
  stage_w(sWhh, Whh, kG, kL, ST64, tid, nth);
  stage_w(sW1, W1, kNH, kL, ST64, tid, nth);
  stage_w(sWm, Wm, kNH, kNH, ST128, tid, nth);
  stage_w(sWf, Wf, kL, kNH, ST128, tid, nth);
  stage_w(sD1, dW1, kNH, kL, ST64, tid, nth);
  stage_w(sD2, dW2, kNH, kNH, ST128, tid, nth);
  for (int i = tid; i < 16 * kNH; i += nth) {            // zero-padded W3^T
    const int n = i / kNH, k = i - n * kNH;
    sD3[n * ST128 + k] = (_Float16)((n < 3) ? dW3[n * kNH + k] : 0.f);
  }
  for (int i = tid; i < kNH; i += nth) {
    sb1[i] = b1[i]; sbm[i] = bm[i]; sdb1[i] = db1[i]; sdb2[i] = db2[i];
  }
  for (int i = tid; i < kL; i += nth) { sbf[i] = bf[i]; sembB[i] = embB[i]; }
  for (int i = tid; i < kG; i += nth) { sbih[i] = bih[i]; sbhh[i] = bhh[i]; }
  for (int i = tid; i < 16; i += nth) sdb3[i] = (i < 3) ? db3[i] : 0.f;
  for (int i = tid; i < kL * 3; i += nth) sembW[i] = embW[i];
  __syncthreads();

  const int wv = tid >> 5, lid = tid & 31;
  const int wtile = blockIdx.x * kWaves + wv;
  if (wtile * 16 >= kB) return;
  const int b0 = wtile * 16;
  _Float16* stg = &stgAll[wv * 16 * SROW];
  float* obsr = &obsrAll[wv * 16 * 3];
  const OdeW ow{sW1, sWm, sWf, sb1, sbm, sbf};

  v8f h[4];
#pragma unroll
  for (int nt = 0; nt < 4; ++nt)
#pragma unroll
    for (int r = 0; r < 8; ++r) h[nt][r] = 0.f;
  float accR = 0.f, accP = 0.f;
  float* pred = out + 5;

#pragma unroll 1
  for (int t = 0; t < kTobs; ++t) {
    const float dt = (t == 0) ? 0.f : (ft[t] - ft[t - 1]);
    if (dt != 0.f) rk4_evolve(h, dt, stg, ow, lid);
    if (lid < 16) {
#pragma unroll
      for (int d = 0; d < 3; ++d)
        obsr[lid * 3 + d] = obs[(b0 + lid) * kTobs * 3 + t * 3 + d];
    }
#pragma unroll 1
    for (int i = lid; i < 16 * kL; i += 32) {            // embedding (K=3, VALU)
      const int row = i >> 6, n = i & 63;
      float v = sembB[n];
#pragma unroll
      for (int d = 0; d < 3; ++d) v += obsr[row * 3 + d] * sembW[n * 3 + d];
      stg[row * SROW + n] = (_Float16)v;
    }
    gru_step(h, stg, sWih, sWhh, sbih, sbhh, lid);
    decode_emit(h, stg, sD1, sD2, sD3, sdb1, sdb2, sdb3, lid, b0, t, pred, obs,
                tgt, accR, accP);
  }
  // tgt_h[0] == h_last (no integration over the first target interval)
  decode_emit(h, stg, sD1, sD2, sD3, sdb1, sdb2, sdb3, lid, b0, kTobs, pred, obs,
              tgt, accR, accP);
#pragma unroll 1
  for (int i = 1; i < kTtgt; ++i) {
    const float dt = ft[kTobs + i] - ft[kTobs + i - 1];
    if (dt != 0.f) rk4_evolve(h, dt, stg, ow, lid);
    decode_emit(h, stg, sD1, sD2, sD3, sdb1, sdb2, sdb3, lid, b0, kTobs + i, pred,
                obs, tgt, accR, accP);
  }
#pragma unroll
  for (int off = 16; off > 0; off >>= 1) {
    accR += __shfl_down(accR, off, 32);
    accP += __shfl_down(accP, off, 32);
  }
  if (lid == 0) { atomicAdd(&ws[0], accR); atomicAdd(&ws[1], accP); }
}

__global__ void ws_init(float* ws) {
  if (threadIdx.x < 2) ws[threadIdx.x] = 0.f;
}

__global__ void finalize_losses(const float* ws, float* out) {
  if (threadIdx.x == 0) {
    const float recon = ws[0] * (1.f / (float)(kB * kTobs * 3));
    const float pl    = ws[1] * (1.f / (float)(kB * kTtgt * 3));
    out[0] = recon + pl; out[1] = recon; out[2] = pl; out[3] = 0.f; out[4] = 0.f;
  }
}

}  // namespace

extern "C" void kernel_launch(void* const* d_in, const int* /*in_sizes*/,
                              int /*n_in*/, void* d_out, int /*out_size*/,
                              void* d_ws, size_t /*ws_size*/, hipStream_t stream) {
  const float* obs  = (const float*)d_in[0];
  const float* tgt  = (const float*)d_in[1];
  const float* ft   = (const float*)d_in[2];
  const float* embW = (const float*)d_in[3];
  const float* embB = (const float*)d_in[4];
  const float* Wih  = (const float*)d_in[5];
  const float* Whh  = (const float*)d_in[6];
  const float* bih  = (const float*)d_in[7];
  const float* bhh  = (const float*)d_in[8];
  const float* W1   = (const float*)d_in[9];
  const float* b1   = (const float*)d_in[10];
  const float* Wm   = (const float*)d_in[11];
  const float* bm   = (const float*)d_in[12];
  const float* Wf   = (const float*)d_in[13];
  const float* bf   = (const float*)d_in[14];
  const float* dW1  = (const float*)d_in[15];
  const float* db1  = (const float*)d_in[16];
  const float* dW2  = (const float*)d_in[17];
  const float* db2  = (const float*)d_in[18];
  const float* dW3  = (const float*)d_in[19];
  const float* db3  = (const float*)d_in[20];
  float* out = (float*)d_out;
  float* ws  = (float*)d_ws;

  ws_init<<<1, 32, 0, stream>>>(ws);
  const int ntiles = kB / 16;  // 1250 wave tiles
  dim3 grid((ntiles + kWaves - 1) / kWaves);
  ode_rnn_fused<<<grid, kWaves * 32, 0, stream>>>(
      obs, tgt, ft, embW, embB, Wih, Whh, bih, bhh, W1, b1, Wm, bm, Wf, bf, dW1,
      db1, dW2, db2, dW3, db3, out, ws);
  finalize_losses<<<1, 32, 0, stream>>>(ws, out);
}
